// Encoder_26706106646720
// MI455X (gfx1250) — matrix operand order repaired and checked by reading, not compiled, not run on hardware
//
#include <hip/hip_runtime.h>

// ---------------------------------------------------------------------------
// Types for CDNA5 WMMA
// ---------------------------------------------------------------------------
typedef __attribute__((ext_vector_type(16))) _Float16 v16h;
typedef __attribute__((ext_vector_type(8)))  float    v8f;
typedef int v4i_ __attribute__((vector_size(16)));

// Problem constants (match reference)
#define Bb 64
#define Tt 512
#define Hh 512
#define Ff 2048
#define Ll 4

// ---------------------------------------------------------------------------
// CDNA5 async global->LDS copy (ASYNCcnt path), with portable fallback
// ---------------------------------------------------------------------------
#if __has_builtin(__builtin_amdgcn_global_load_async_to_lds_b128)
#define HAS_ASYNC_LDS 1
#else
#define HAS_ASYNC_LDS 0
#endif

__device__ __forceinline__ void cp16_lds(void* lds_dst, const void* gsrc) {
#if HAS_ASYNC_LDS
    typedef __attribute__((address_space(1))) v4i_ GVec;  // AS1 = __device__
    typedef __attribute__((address_space(3))) v4i_ LVec;  // AS3 = LDS
    __builtin_amdgcn_global_load_async_to_lds_b128(
        (GVec*)(uintptr_t)gsrc,
        (LVec*)(unsigned)(uintptr_t)lds_dst,
        0, 0);
#else
    *(float4*)lds_dst = *(const float4*)gsrc;
#endif
}

template<int N>
__device__ __forceinline__ void async_wait() {
#if HAS_ASYNC_LDS
#if __has_builtin(__builtin_amdgcn_s_wait_asynccnt)
    __builtin_amdgcn_s_wait_asynccnt(N);
#else
    asm volatile("s_wait_asynccnt %0" :: "n"(N));
#endif
#endif
}

// ---------------------------------------------------------------------------
// Fast activations
// ---------------------------------------------------------------------------
__device__ __forceinline__ float sig_f(float x) {
    return 1.0f / (1.0f + __expf(-x));
}
__device__ __forceinline__ float tanh_f(float x) {
    // stable: tanh(x) = 1 - 2/(1+exp(2x))
    return 1.0f - 2.0f / (1.0f + __expf(2.0f * x));
}

// ---------------------------------------------------------------------------
// Conversion helpers (run once per layer / once per launch; negligible cost)
// ---------------------------------------------------------------------------
__global__ __launch_bounds__(256)
void cvt_f32_to_f16_kernel(const float* __restrict__ in,
                           _Float16* __restrict__ out) {
    const size_t i = (size_t)blockIdx.x * 256 + threadIdx.x;
    out[i] = (_Float16)in[i];
}

// out[n*K + k] = (f16) in[k*N + n]   (weights -> f16, transposed to [N][K])
__global__ __launch_bounds__(256)
void cvt_transpose_kernel(const float* __restrict__ in,
                          _Float16* __restrict__ out, int K, int N) {
    const size_t i = (size_t)blockIdx.x * 256 + threadIdx.x;
    const int k = (int)(i % K);
    const int n = (int)(i / K);
    out[(size_t)n * K + k] = (_Float16)in[(size_t)k * N + n];
}

// ---------------------------------------------------------------------------
// Tiled WMMA GEMM:  C[M,N] = op(A[M,K] * B^T  + bias[N])
//   A:  f16 row-major [M][K]
//   Bt: f16 row-major [N][K]  (pre-transposed weights)
//   Output: f32 or f16 (OUT_HALF), optional ReLU
// Tile: 128x128x32, 256 threads = 8 waves; wave does 32(M)x64(N) = 2x4 tiles.
// Double-buffered LDS, async global->LDS staging (ASYNCcnt pipelined).
// ---------------------------------------------------------------------------
#define AS 40  // LDS row stride in halfs (80B, 16B-aligned)

template<bool OUT_HALF, bool RELU>
__global__ __launch_bounds__(256)
void gemm_wmma_kernel(const _Float16* __restrict__ Ah,
                      const _Float16* __restrict__ Bt,
                      const float* __restrict__ bias,
                      void* __restrict__ Cp,
                      int M, int N, int K) {
    __shared__ _Float16 a_smem[2][128 * AS];
    __shared__ _Float16 b_smem[2][128 * AS];

    const int tid   = threadIdx.x;
    const int lane  = tid & 31;
    const int wave  = tid >> 5;
    const int half_ = lane >> 4;
    const int ln    = lane & 15;

    const int m0 = blockIdx.y * 128;
    const int n0 = blockIdx.x * 128;
    const int mbase = (wave & 3) * 32;   // wave's M sub-block
    const int nbase = (wave >> 2) * 64;  // wave's N sub-block

    v8f acc[2][4];
#pragma unroll
    for (int mt = 0; mt < 2; ++mt)
#pragma unroll
        for (int nt = 0; nt < 4; ++nt)
#pragma unroll
            for (int v = 0; v < 8; ++v) acc[mt][nt][v] = 0.0f;

    // stage one 128x32 f16 tile of A and of Bt into LDS buffer p (b128 copies)
    auto issue_tile = [&](int p, int k0) {
        const int i0 = tid * 2;
#pragma unroll
        for (int q = 0; q < 2; ++q) {
            const int i   = i0 + q;
            const int row = i >> 2;        // 0..127
            const int seg = (i & 3) * 8;   // 0,8,16,24 halfs
            cp16_lds(&a_smem[p][row * AS + seg],
                     Ah + (size_t)(m0 + row) * K + k0 + seg);
            cp16_lds(&b_smem[p][row * AS + seg],
                     Bt + (size_t)(n0 + row) * K + k0 + seg);
        }
    };

    const int KT = K >> 5;
    issue_tile(0, 0);

    for (int kb = 0; kb < KT; ++kb) {
        const int p = kb & 1;
        async_wait<0>();      // tile kb resident in LDS (this wave's copies)
        __syncthreads();      // all waves' copies visible; prev compute done
        if (kb + 1 < KT) issue_tile(p ^ 1, (kb + 1) * 32);

        // ---- build fragments (documented CDNA5 16-bit layouts) ----
        v16h afr[2];
#pragma unroll
        for (int mt = 0; mt < 2; ++mt) {
            const int m = mbase + mt * 16 + ln;
            union { v16h v; unsigned u[8]; } U;
#pragma unroll
            for (int j = 0; j < 8; ++j) {
                const int k2 = 16 * (j >> 2) + 8 * half_ + 2 * (j & 3);
                U.u[j] = *(const unsigned*)&a_smem[p][m * AS + k2];
            }
            afr[mt] = U.v;
        }
        v16h bfr[4];
#pragma unroll
        for (int nt = 0; nt < 4; ++nt) {
            const int n = nbase + nt * 16 + ln;
            union { v16h v; unsigned u[8]; } U;
#pragma unroll
            for (int j = 0; j < 8; ++j) {
                const int k2 = 16 * (j >> 2) + 8 * half_ + 2 * (j & 3);
                U.u[j] = *(const unsigned*)&b_smem[p][n * AS + k2];
            }
            bfr[nt] = U.v;
        }

#pragma unroll
        for (int mt = 0; mt < 2; ++mt)
#pragma unroll
            for (int nt = 0; nt < 4; ++nt)
                acc[mt][nt] = __builtin_amdgcn_wmma_f32_16x16x32_f16(
                    false, afr[mt], false, bfr[nt], (short)0, acc[mt][nt],
                    false, false);
    }

    // ---- epilogue: bias (+ReLU) and store (C-layout: m = 8*half + v) ----
#pragma unroll
    for (int mt = 0; mt < 2; ++mt)
#pragma unroll
        for (int nt = 0; nt < 4; ++nt) {
            const int n  = n0 + nbase + nt * 16 + ln;
            const float bv = bias[n];
#pragma unroll
            for (int v = 0; v < 8; ++v) {
                const int m = m0 + mbase + mt * 16 + 8 * half_ + v;
                float val = acc[mt][nt][v] + bv;
                if (RELU) val = fmaxf(val, 0.0f);
                if (OUT_HALF)
                    ((_Float16*)Cp)[(size_t)m * N + n] = (_Float16)val;
                else
                    ((float*)Cp)[(size_t)m * N + n] = val;
            }
        }
}

// ---------------------------------------------------------------------------
// Persistent LSTM scan kernel.
// Grid = 16 WGs x 256 threads (8 waves). WG owns 32 hidden units (jb..jb+31).
// Wht slice (f16, pre-transposed [4H][H], 135 KB) and h_prev (64x512 f16,
// 68 KB) live in dynamic LDS (~200 KB of the WGP's 320 KB), filled with
// async global->LDS copies. Wave w: batch tile mt=w&3, hidden half sub=w>>2
// -> computes i/f/g/o 16x16 tiles and keeps cell state c in registers.
// Device-wide release/acquire barrier per timestep.
// ---------------------------------------------------------------------------
#define KS 528  // LDS k-stride in halfs (1056B, 16B-aligned)
#define NWG 16

__global__ __launch_bounds__(256)
void lstm_scan_kernel(const float* __restrict__ P,        // [B,T,4H] pregates f32
                      const _Float16* __restrict__ Wht,   // [4H][H] f16 (transposed)
                      _Float16* __restrict__ Hid,         // [B,T,H] f16 out
                      _Float16* __restrict__ Hprev,       // [2][B,H] f16 ping-pong
                      unsigned* __restrict__ syncCnt,
                      int rev) {
    extern __shared__ _Float16 smem[];
    _Float16* whs = smem;                 // [128][KS]
    _Float16* hs  = smem + 128 * KS;      // [64][KS]

    const int tid   = threadIdx.x;
    const int lane  = tid & 31;
    const int wave  = tid >> 5;
    const int half_ = lane >> 4;
    const int ln    = lane & 15;
    const int jb    = blockIdx.x * 32;    // hidden base of this WG
    const int mt    = wave & 3;           // batch tile (16 rows)
    const int sub   = wave >> 2;          // hidden sub-slice (0/1)

    // Preload Wh slice: whs[nl][k], nl = g*32 + local col; async b128 copies.
    for (int idx = tid; idx < 128 * 64; idx += 256) {
        const int nl  = idx >> 6;        // 0..127
        const int seg = (idx & 63) * 8;  // 0..504
        const int g   = nl >> 5;
        const int col = g * 512 + jb + (nl & 31);
        cp16_lds(&whs[nl * KS + seg], Wht + (size_t)col * Hh + seg);
    }

    v8f c;
#pragma unroll
    for (int v = 0; v < 8; ++v) c[v] = 0.0f;

    const int jcol = jb + sub * 16 + ln;  // this lane's global hidden column

    for (int step = 0; step < Tt; ++step) {
        const int t = rev ? (Tt - 1 - step) : step;

        // stage h_{t-1} (f16, [64][512]) into LDS via async b128
        const _Float16* hp = Hprev + (size_t)(step & 1) * (Bb * Hh);
        for (int idx = tid; idx < (Bb * Hh / 8); idx += 256) {
            const int off = idx * 8;
            const int m = off >> 9;
            const int k = off & 511;
            cp16_lds(&hs[m * KS + k], hp + off);
        }

        // prefetch this step's pregate rows while the recurrent GEMM runs
#pragma unroll
        for (int v = 0; v < 8; ++v) {
            const int m = mt * 16 + 8 * half_ + v;
            const float* pb = P + ((size_t)m * Tt + t) * (4 * Hh) + jcol;
            __builtin_prefetch(pb,        0, 1);
            __builtin_prefetch(pb +  512, 0, 1);
            __builtin_prefetch(pb + 1024, 0, 1);
            __builtin_prefetch(pb + 1536, 0, 1);
        }

        async_wait<0>();   // whs (step 0) + hs copies complete
        __syncthreads();

        v8f acc[4];
#pragma unroll
        for (int g = 0; g < 4; ++g)
#pragma unroll
            for (int v = 0; v < 8; ++v) acc[g][v] = 0.0f;

        const int m_a = mt * 16 + ln;
#pragma unroll 4
        for (int kb = 0; kb < 16; ++kb) {
            const int kbase = kb * 32;
            union { v16h v; unsigned u[8]; } A;
#pragma unroll
            for (int j = 0; j < 8; ++j) {
                const int k2 = kbase + 16 * (j >> 2) + 8 * half_ + 2 * (j & 3);
                A.u[j] = *(const unsigned*)&hs[m_a * KS + k2];
            }
#pragma unroll
            for (int g = 0; g < 4; ++g) {
                const int nl = g * 32 + sub * 16 + ln;
                union { v16h v; unsigned u[8]; } Bf;
#pragma unroll
                for (int j = 0; j < 8; ++j) {
                    const int k2 = kbase + 16 * (j >> 2) + 8 * half_ + 2 * (j & 3);
                    Bf.u[j] = *(const unsigned*)&whs[nl * KS + k2];
                }
                acc[g] = __builtin_amdgcn_wmma_f32_16x16x32_f16(
                    false, A.v, false, Bf.v, (short)0, acc[g], false, false);
            }
        }

        // elementwise gates + state update (all in WMMA C-layout registers)
        _Float16* hpn = Hprev + (size_t)((step + 1) & 1) * (Bb * Hh);
#pragma unroll
        for (int v = 0; v < 8; ++v) {
            const int m = mt * 16 + 8 * half_ + v;
            const size_t prow = ((size_t)m * Tt + t) * (4 * Hh);
            const float gi = acc[0][v] + P[prow + 0 * Hh + jcol];
            const float gf = acc[1][v] + P[prow + 1 * Hh + jcol];
            const float gg = acc[2][v] + P[prow + 2 * Hh + jcol];
            const float go = acc[3][v] + P[prow + 3 * Hh + jcol];
            const float cv = sig_f(gf) * c[v] + sig_f(gi) * tanh_f(gg);
            c[v] = cv;
            const float hv = sig_f(go) * tanh_f(cv);
            Hid[((size_t)m * Tt + t) * Hh + jcol] = (_Float16)hv;
            hpn[(size_t)m * Hh + jcol] = (_Float16)hv;
        }

        // device-wide release/acquire barrier between timesteps
        __threadfence();
        __syncthreads();
        if (tid == 0) {
            atomicAdd(syncCnt, 1u);
            const unsigned target = (unsigned)NWG * (unsigned)(step + 1);
            while (__hip_atomic_load(syncCnt, __ATOMIC_RELAXED,
                                     __HIP_MEMORY_SCOPE_AGENT) < target)
                __builtin_amdgcn_s_sleep(1);
        }
        __syncthreads();
        __threadfence();
    }
}

// ---------------------------------------------------------------------------
// Workspace layout (bytes)
// ---------------------------------------------------------------------------
#define OFF_P    ((size_t)0)                          // [B,T,4H] f32 = 256 MB
#define OFF_Z    (OFF_P    + (size_t)Bb*Tt*4*Hh*4)    // [B,T,F]  f16 = 128 MB
#define OFF_HID  (OFF_Z    + (size_t)Bb*Tt*Ff*2)      // [B,T,H]  f16 =  32 MB
#define OFF_XH   (OFF_HID  + (size_t)Bb*Tt*Hh*2)      // [B,T,H]  f16 =  32 MB
#define OFF_WXT  (OFF_XH   + (size_t)Bb*Tt*Hh*2)      // [4H][H]  f16 =   2 MB
#define OFF_WHT  (OFF_WXT  + (size_t)Hh*4*Hh*2)       // [4H][H]  f16
#define OFF_W1T  (OFF_WHT  + (size_t)Hh*4*Hh*2)       // [F][H]   f16
#define OFF_W2T  (OFF_W1T  + (size_t)Hh*Ff*2)         // [H][F]   f16
#define OFF_HP   (OFF_W2T  + (size_t)Ff*Hh*2)         // 2x[B,H]  f16
#define OFF_SYNC (OFF_HP   + (size_t)2*Bb*Hh*2)

extern "C" void kernel_launch(void* const* d_in, const int* in_sizes, int n_in,
                              void* d_out, int out_size, void* d_ws, size_t ws_size,
                              hipStream_t stream) {
    (void)in_sizes; (void)n_in; (void)out_size; (void)ws_size;

    const float* x  = (const float*)d_in[0];
    const float* Wx = (const float*)d_in[1];
    const float* Wh = (const float*)d_in[2];
    const float* b  = (const float*)d_in[3];
    const float* W1 = (const float*)d_in[4];
    const float* b1 = (const float*)d_in[5];
    const float* W2 = (const float*)d_in[6];
    const float* b2 = (const float*)d_in[7];

    char* ws = (char*)d_ws;
    float*     P    = (float*)(ws + OFF_P);
    _Float16*  Z    = (_Float16*)(ws + OFF_Z);
    _Float16*  Hid  = (_Float16*)(ws + OFF_HID);
    _Float16*  Xh   = (_Float16*)(ws + OFF_XH);
    _Float16*  Wxt  = (_Float16*)(ws + OFF_WXT);
    _Float16*  Wht  = (_Float16*)(ws + OFF_WHT);
    _Float16*  W1t  = (_Float16*)(ws + OFF_W1T);
    _Float16*  W2t  = (_Float16*)(ws + OFF_W2T);
    _Float16*  Hp   = (_Float16*)(ws + OFF_HP);
    unsigned*  sync = (unsigned*)(ws + OFF_SYNC);

    const int M = Bb * Tt;                       // 32768
    const int lstm_smem = (128 + 64) * KS * 2;   // ~203 KB dynamic LDS

    (void)hipFuncSetAttribute(reinterpret_cast<const void*>(lstm_scan_kernel),
                              hipFuncAttributeMaxDynamicSharedMemorySize,
                              lstm_smem);

    // input activations -> f16 once
    cvt_f32_to_f16_kernel<<<(M * Hh) / 256, 256, 0, stream>>>(x, Xh);

    for (int l = 0; l < Ll; ++l) {
        const int rev = (l & 1);

        // per-layer weights -> f16, transposed to [N][K]
        cvt_transpose_kernel<<<(Hh * 4 * Hh) / 256, 256, 0, stream>>>(
            Wx + (size_t)l * Hh * 4 * Hh, Wxt, Hh, 4 * Hh);
        cvt_transpose_kernel<<<(Hh * 4 * Hh) / 256, 256, 0, stream>>>(
            Wh + (size_t)l * Hh * 4 * Hh, Wht, Hh, 4 * Hh);
        cvt_transpose_kernel<<<(Hh * Ff) / 256, 256, 0, stream>>>(
            W1 + (size_t)l * Hh * Ff, W1t, Hh, Ff);
        cvt_transpose_kernel<<<(Ff * Hh) / 256, 256, 0, stream>>>(
            W2 + (size_t)l * Ff * Hh, W2t, Ff, Hh);

        (void)hipMemsetAsync(Hp, 0, (size_t)2 * Bb * Hh * 2, stream);
        (void)hipMemsetAsync(sync, 0, 256, stream);

        // 1) pregates: P = Xh @ Wx[l]^T' + b[l]   (M x 2048, K=512, f32 out)
        gemm_wmma_kernel<false, false>
            <<<dim3(4 * Hh / 128, M / 128), 256, 0, stream>>>(
                Xh, Wxt, b + (size_t)l * 4 * Hh, P, M, 4 * Hh, Hh);

        // 2) recurrent scan (reverse time order on odd layers)
        lstm_scan_kernel<<<NWG, 256, lstm_smem, stream>>>(
            P, Wht, Hid, Hp, sync, rev);

        // 3) FFN1: Z = relu(Hid @ W1 + b1) -> f16  (M x 2048, K=512)
        gemm_wmma_kernel<true, true>
            <<<dim3(Ff / 128, M / 128), 256, 0, stream>>>(
                Hid, W1t, b1 + (size_t)l * Ff, Z, M, Ff, Hh);

        // 4) FFN2: X/out = Z @ W2 + b2   (M x 512, K=2048)
        if (l == Ll - 1) {
            gemm_wmma_kernel<false, false>
                <<<dim3(Hh / 128, M / 128), 256, 0, stream>>>(
                    Z, W2t, b2 + (size_t)l * Hh, (float*)d_out, M, Hh, Ff);
        } else {
            gemm_wmma_kernel<true, false>
                <<<dim3(Hh / 128, M / 128), 256, 0, stream>>>(
                    Z, W2t, b2 + (size_t)l * Hh, Xh, M, Hh, Ff);
        }
    }
}
